// MaskedSingleHeadAttention_42296837931766
// MI455X (gfx1250) — matrix operand order repaired and checked by reading, not compiled, not run on hardware
//
#include <hip/hip_runtime.h>

// ---------------------------------------------------------------------------
// MaskedSingleHeadAttention on MI455X (gfx1250, wave32, WMMA + TDM).
// Pass 0: f32 -> bf16 conversion of x / Wq / Wk / Wv (packed converts).
// Pass 1: fused QKV projection via v_wmma_f32_16x16x32_bf16,
//         A-tile double-buffered through the Tensor Data Mover.
// Pass 2: per-16-row causal attention; score row in LDS; Q via TDM;
//         V chunks ping-pong TDM buffers overlaid on the dead score buffer.
// ---------------------------------------------------------------------------

typedef __attribute__((ext_vector_type(16))) __bf16 v16bf;
typedef __attribute__((ext_vector_type(8)))  __bf16 v8bf;
typedef __attribute__((ext_vector_type(8)))  float  v8ff;
typedef __attribute__((ext_vector_type(8)))  float  v8f;

#define WMMA_BF16(a, b, c) \
    __builtin_amdgcn_wmma_f32_16x16x32_bf16(false, (a), false, (b), (short)0, (c), false, false)

#define D_MODEL 1024
#define SEQ     2048
#define BATCH   4
#define NROWS   (BATCH * SEQ)      // 8192

// ===========================================================================
// Tensor Data Mover: 2D bf16 tile load, global -> LDS (D# per ISA ch.8).
// ===========================================================================
#if __has_builtin(__builtin_amdgcn_tensor_load_to_lds)
#define HAVE_TDM 1
typedef unsigned int u32x4 __attribute__((ext_vector_type(4)));
typedef int          i32x4 __attribute__((ext_vector_type(4)));
typedef int          i32x8 __attribute__((ext_vector_type(8)));

__device__ __forceinline__
void tdm_load_2d_bf16(unsigned lds_off, const void* gptr,
                      unsigned tile_d0, unsigned tile_d1,
                      unsigned tensor_d0, unsigned tensor_d1,
                      unsigned d0_stride)
{
    unsigned long long ga = (unsigned long long)(uintptr_t)gptr;
    u32x4 g0;
    g0.x = 1u;                                   // count=1, user descriptor
    g0.y = lds_off;                              // lds_addr (bytes)
    g0.z = (unsigned)(ga & 0xffffffffull);       // global_addr[31:0]
    g0.w = (unsigned)((ga >> 32) & 0x1ffffffull) // global_addr[56:32]
         | (2u << 30);                           // type = 2 ("image")
    i32x8 g1;
    g1[0] = (int)(1u << 16);                     // data_size=1 (2 bytes)
    g1[1] = (int)((tensor_d0 & 0xffffu) << 16);  // tensor_dim0[15:0] @ bits 63:48
    g1[2] = (int)(((tensor_d0 >> 16) & 0xffffu) | ((tensor_d1 & 0xffffu) << 16));
    g1[3] = (int)(((tensor_d1 >> 16) & 0xffffu) | ((tile_d0 & 0xffffu) << 16));
    g1[4] = (int)(tile_d1 & 0xffffu);            // tile_dim1; tile_dim2=0
    g1[5] = (int)d0_stride;                      // tensor_dim0_stride[31:0]
    g1[6] = 0;
    g1[7] = 0;
    i32x4 z4 = {0, 0, 0, 0};
#if defined(__clang_major__) && (__clang_major__ >= 23)
    i32x8 z8 = {0, 0, 0, 0, 0, 0, 0, 0};
    __builtin_amdgcn_tensor_load_to_lds(g0, g1, z4, z4, z8, 0);
#else
    __builtin_amdgcn_tensor_load_to_lds(g0, g1, z4, z4, 0);
#endif
}
#endif

__device__ __forceinline__ void wait_tensorcnt0()
{
#if __has_builtin(__builtin_amdgcn_s_wait_tensorcnt)
    __builtin_amdgcn_s_wait_tensorcnt(0);
#else
    asm volatile("s_wait_tensorcnt 0x0" ::: "memory");
#endif
}

// ===========================================================================
// Pass 0: packed f32 -> bf16 conversion (one-time; weights/x reused many x).
// ===========================================================================
__global__ __launch_bounds__(256)
void cvt_bf16_kernel(const float* __restrict__ src, __bf16* __restrict__ dst)
{
    const size_t i = ((size_t)blockIdx.x * 256 + threadIdx.x) * 8;
    v8ff f = *(const v8ff*)(src + i);
    *(v8bf*)(dst + i) = __builtin_convertvector(f, v8bf);
}

// ===========================================================================
// Pass 1: fused QKV projection, bf16 in -> bf16 out (+f32 bias).
//   128x128 tile / workgroup; K tiled by 32; double-buffered A via TDM.
// ===========================================================================
#define L1_AS0 0                        // [128][32] bf16 (8192 B)
#define L1_AS1 8192
#define L1_WT0 16384                    // [128][34] bf16 transposed [n][k]
#define L1_WT1 25088
#define L1_TOT (25088 + 128 * 34 * 2)   // 33792

__device__ __forceinline__
void stage_w_tile(const __bf16* __restrict__ W, int k0, int col0,
                  __bf16* __restrict__ Wt, int tid)
{
    const int kk = tid >> 3;
    const int c0 = (tid & 7) * 16;
    const v16bf u = *(const v16bf*)&W[(size_t)(k0 + kk) * D_MODEL + col0 + c0];
#pragma unroll
    for (int e = 0; e < 16; ++e)
        Wt[(c0 + e) * 34 + kk] = u[e];
}

#if !defined(HAVE_TDM)
__device__ __forceinline__
void stage_a_tile(const __bf16* __restrict__ xb, int row0, int k0,
                  __bf16* __restrict__ As, int tid)
{
    const int r = tid >> 1, h = (tid & 1) * 16;
    *(v16bf*)&As[r * 32 + h] =
        *(const v16bf*)&xb[(size_t)(row0 + r) * D_MODEL + k0 + h];
}
#endif

__global__ __launch_bounds__(256)
void qkv_proj_kernel(const __bf16* __restrict__ xb,
                     const __bf16* __restrict__ Wqb, const float* __restrict__ bq,
                     const __bf16* __restrict__ Wkb, const float* __restrict__ bk,
                     const __bf16* __restrict__ Wvb, const float* __restrict__ bv,
                     __bf16* __restrict__ Qb, __bf16* __restrict__ Kb,
                     __bf16* __restrict__ Vb)
{
    const __bf16* W;  const float* bias;  __bf16* out;
    if (blockIdx.z == 0)      { W = Wqb; bias = bq; out = Qb; }
    else if (blockIdx.z == 1) { W = Wkb; bias = bk; out = Kb; }
    else                      { W = Wvb; bias = bv; out = Vb; }

    __shared__ char lds1[L1_TOT];

    const int tid  = threadIdx.x;
    const int wave = tid >> 5;
    const int lane = tid & 31;
    const int ln   = lane & 15;
    const int hi   = lane >> 4;

    const int row0 = blockIdx.x * 128;
    const int col0 = blockIdx.y * 128;

    v8f acc[8];
    const v8f vzero = {0.f, 0.f, 0.f, 0.f, 0.f, 0.f, 0.f, 0.f};
#pragma unroll
    for (int j = 0; j < 8; ++j) acc[j] = vzero;

    // ---- prologue: fill buffer 0 ----
#if HAVE_TDM
    if (wave == 0)
        tdm_load_2d_bf16(L1_AS0, &xb[(size_t)row0 * D_MODEL],
                         32, 128, D_MODEL, NROWS, D_MODEL);
#else
    stage_a_tile(xb, row0, 0, (__bf16*)(lds1 + L1_AS0), tid);
#endif
    stage_w_tile(W, 0, col0, (__bf16*)(lds1 + L1_WT0), tid);
#if HAVE_TDM
    if (wave == 0) wait_tensorcnt0();
#endif
    __syncthreads();

    for (int k0 = 0; k0 < D_MODEL; k0 += 32) {
        const int cur = (k0 >> 5) & 1;
        __bf16* As = (__bf16*)(lds1 + (cur ? L1_AS1 : L1_AS0));
        __bf16* Wt = (__bf16*)(lds1 + (cur ? L1_WT1 : L1_WT0));

        // ---- issue next tile into the other buffer (overlaps WMMA) ----
        const int k1 = k0 + 32;
        if (k1 < D_MODEL) {
#if HAVE_TDM
            if (wave == 0)
                tdm_load_2d_bf16(cur ? L1_AS0 : L1_AS1,
                                 &xb[(size_t)row0 * D_MODEL + k1],
                                 32, 128, D_MODEL, NROWS, D_MODEL);
#else
            stage_a_tile(xb, row0, k1,
                         (__bf16*)(lds1 + (cur ? L1_AS0 : L1_AS1)), tid);
#endif
            stage_w_tile(W, k1, col0,
                         (__bf16*)(lds1 + (cur ? L1_WT0 : L1_WT1)), tid);
        }

        // ---- A fragment (16x32 bf16, k = 16*hi + s, same order as B) ----
        v16bf a;
        {
            const int r = (wave << 4) + ln;
#pragma unroll
            for (int v = 0; v < 8; ++v) {
                const int kb = (hi << 4) + (v << 1);
                a[2 * v]     = As[r * 32 + kb];
                a[2 * v + 1] = As[r * 32 + kb + 1];
            }
        }
#pragma unroll
        for (int j = 0; j < 8; ++j) {
            const int c = (j << 4) + ln;
            v16bf bfrag;
#pragma unroll
            for (int v = 0; v < 8; ++v) {
                const int kb = (hi << 4) + (v << 1);
                bfrag[2 * v]     = Wt[c * 34 + kb];
                bfrag[2 * v + 1] = Wt[c * 34 + kb + 1];
            }
            acc[j] = WMMA_BF16(a, bfrag, acc[j]);
        }
#if HAVE_TDM
        if (wave == 0) wait_tensorcnt0();
#endif
        __syncthreads();
    }

    // ---- epilogue: +bias (f32), store bf16 ----
    const int mrow0 = row0 + (wave << 4) + (hi << 3);
#pragma unroll
    for (int j = 0; j < 8; ++j) {
        const int c  = col0 + (j << 4) + ln;
        const float bb = bias[c];
#pragma unroll
        for (int r = 0; r < 8; ++r)
            out[(size_t)(mrow0 + r) * D_MODEL + c] = (__bf16)(acc[j][r] + bb);
    }
}

// ===========================================================================
// Pass 2: causal attention, one 16-query-row tile per workgroup.
//   LDS: Qs 16x1024 bf16 | Ss 16x2048 f32 (reused as V ping-pong after
//   softmax) | Ps 16x2048 bf16 | reductions.  Total 230.5 KB (< 320 KB).
// ===========================================================================
#define LDS_QS   0
#define LDS_SS   32768                       // scores, dead after softmax
#define LDS_VS0  32768                       // V buf 0 overlays Ss[  0..64K)
#define LDS_VS1  98304                       // V buf 1 overlays Ss[64K..128K)
#define LDS_PS   (32768 + 131072)            // 163840
#define LDS_RED  (163840 + 65536)            // 229376
#define LDS_TOT  (229376 + 256 * 4 + 32 * 4) // 230528

__global__ __launch_bounds__(256)
void attn_kernel(const __bf16* __restrict__ Qb,
                 const __bf16* __restrict__ Kb,
                 const __bf16* __restrict__ Vb,
                 float* __restrict__ out)
{
    extern __shared__ char smem[];
    __bf16* Qs     = (__bf16*)(smem + LDS_QS);   // [16][1024]
    float*  Ss     = (float*) (smem + LDS_SS);   // [16][2048]
    __bf16* Ps     = (__bf16*)(smem + LDS_PS);   // [16][2048]
    float*  red    = (float*) (smem + LDS_RED);  // [16][16]
    float*  rowmax = red + 256;
    float*  rowsum = rowmax + 16;

    const int tid  = threadIdx.x;
    const int wave = tid >> 5;
    const int lane = tid & 31;
    const int ln   = lane & 15;
    const int hi   = lane >> 4;

    const int b  = blockIdx.x / (SEQ / 16);
    const int qt = blockIdx.x % (SEQ / 16);
    const int q0 = qt * 16;
    const size_t base = (size_t)b * SEQ;

    // ---- Q tile 16x1024 into LDS via TDM ----
#if HAVE_TDM
    if (wave == 0) {
        tdm_load_2d_bf16(LDS_QS, &Qb[(base + q0) * D_MODEL],
                         D_MODEL, 16, D_MODEL, NROWS, D_MODEL);
        wait_tensorcnt0();
    }
#else
    {
        const int r  = tid >> 4;
        const int c0 = (tid & 15) * 64;
#pragma unroll
        for (int i = 0; i < 4; ++i)
            *(v16bf*)&Qs[r * D_MODEL + c0 + i * 16] =
                *(const v16bf*)&Qb[(base + q0 + r) * D_MODEL + c0 + i * 16];
    }
#endif
    __syncthreads();

    const float scale = 0.03125f;   // 1/sqrt(1024)

    // ---- scores: wave owns keys [wave*256, +256), 4 groups of 64 ----
    for (int g = 0; g < 4; ++g) {
        const int kg0 = wave * 256 + g * 64;
        if (kg0 > q0 + 15) {
            for (int i = lane; i < 16 * 64; i += 32) {
                const int rr = i >> 6, cc = i & 63;
                Ss[rr * 2048 + kg0 + cc] = -__builtin_inff();
            }
            continue;
        }
        v8f acc[4];
        const v8f vzero = {0.f, 0.f, 0.f, 0.f, 0.f, 0.f, 0.f, 0.f};
#pragma unroll
        for (int t = 0; t < 4; ++t) acc[t] = vzero;

        for (int d0 = 0; d0 < D_MODEL; d0 += 32) {
            __builtin_prefetch(&Kb[(base + kg0 + ln) * D_MODEL + d0 + 256], 0, 1);
            v16bf a;
#pragma unroll
            for (int v = 0; v < 8; ++v) {
                const int kb = (hi << 4) + (v << 1);
                a[2 * v]     = Qs[ln * D_MODEL + d0 + kb];
                a[2 * v + 1] = Qs[ln * D_MODEL + d0 + kb + 1];
            }
#pragma unroll
            for (int t = 0; t < 4; ++t) {
                const int krow = kg0 + t * 16 + ln;
                const v16bf bfrag =
                    *(const v16bf*)&Kb[(base + krow) * D_MODEL + d0 + hi * 16];
                acc[t] = WMMA_BF16(a, bfrag, acc[t]);
            }
        }
#pragma unroll
        for (int t = 0; t < 4; ++t) {
            const int kcol = kg0 + t * 16 + ln;
#pragma unroll
            for (int r = 0; r < 8; ++r) {
                const int m = (hi << 3) + r;
                Ss[m * 2048 + kcol] =
                    (kcol <= q0 + m) ? acc[t][r] * scale : -__builtin_inff();
            }
        }
    }
    __syncthreads();

    // ---- softmax: 16 threads/row, 128 elems each, vectorized LDS access ----
    {
        const int r   = tid >> 4;
        const int seg = tid & 15;
        float* srow = &Ss[r * 2048 + seg * 128];
        float mx = -__builtin_inff();
        for (int i = 0; i < 128; i += 8) {
            v8ff s = *(const v8ff*)&srow[i];
#pragma unroll
            for (int e = 0; e < 8; ++e) mx = fmaxf(mx, s[e]);
        }
        red[r * 16 + seg] = mx;
        __syncthreads();
        if (seg == 0) {
            float m2 = -__builtin_inff();
#pragma unroll
            for (int i = 0; i < 16; ++i) m2 = fmaxf(m2, red[r * 16 + i]);
            rowmax[r] = m2;
        }
        __syncthreads();
        const float rm = rowmax[r];
        __bf16* prow = &Ps[r * 2048 + seg * 128];
        float sum = 0.f;
        for (int i = 0; i < 128; i += 8) {
            v8ff s = *(const v8ff*)&srow[i];
            v8ff e;
#pragma unroll
            for (int k = 0; k < 8; ++k) { e[k] = __expf(s[k] - rm); sum += e[k]; }
            *(v8bf*)&prow[i] = __builtin_convertvector(e, v8bf);
        }
        red[r * 16 + seg] = sum;
        __syncthreads();
        if (seg == 0) {
            float s2 = 0.f;
#pragma unroll
            for (int i = 0; i < 16; ++i) s2 += red[r * 16 + i];
            rowsum[r] = s2;
        }
        __syncthreads();
        // Ss is now dead: its 128KB become the V ping-pong buffers.
    }

    // ---- O = P @ V : wave owns cols [wave*128, +128);
    //      V chunks double-buffered through TDM over the dead Ss region ----
    v8f oacc[8];
    {
        const v8f vzero = {0.f, 0.f, 0.f, 0.f, 0.f, 0.f, 0.f, 0.f};
#pragma unroll
        for (int j = 0; j < 8; ++j) oacc[j] = vzero;
    }
    const int nkc = (q0 + 16 + 31) >> 5;   // causal: only keys <= q0+15

    // prologue: chunk 0 into buffer 0
#if HAVE_TDM
    if (wave == 0) {
        tdm_load_2d_bf16(LDS_VS0, &Vb[base * D_MODEL],
                         D_MODEL, 32, D_MODEL, NROWS, D_MODEL);
        wait_tensorcnt0();
    }
#else
    {
        const int r  = tid >> 3;
        const int c0 = (tid & 7) * 128;
        __bf16* Vs = (__bf16*)(smem + LDS_VS0);
#pragma unroll
        for (int i = 0; i < 8; ++i)
            *(v16bf*)&Vs[r * D_MODEL + c0 + i * 16] =
                *(const v16bf*)&Vb[(base + r) * D_MODEL + c0 + i * 16];
    }
#endif
    __syncthreads();

    for (int kc = 0; kc < nkc; ++kc) {
        __bf16* Vs = (__bf16*)(smem + ((kc & 1) ? LDS_VS1 : LDS_VS0));

        // issue next chunk into the other buffer (overlaps WMMA below)
        if (kc + 1 < nkc) {
#if HAVE_TDM
            if (wave == 0)
                tdm_load_2d_bf16((kc & 1) ? LDS_VS0 : LDS_VS1,
                                 &Vb[(base + (size_t)(kc + 1) * 32) * D_MODEL],
                                 D_MODEL, 32, D_MODEL, NROWS, D_MODEL);
#else
            {
                const int r  = tid >> 3;
                const int c0 = (tid & 7) * 128;
                __bf16* Vn = (__bf16*)(smem + ((kc & 1) ? LDS_VS0 : LDS_VS1));
#pragma unroll
                for (int i = 0; i < 8; ++i)
                    *(v16bf*)&Vn[r * D_MODEL + c0 + i * 16] =
                        *(const v16bf*)&Vb[(base + (size_t)(kc + 1) * 32 + r) * D_MODEL
                                           + c0 + i * 16];
            }
#endif
        }

        const int kk = kc * 32;
        v16bf a;
#pragma unroll
        for (int v = 0; v < 8; ++v) {
            const int kb = (hi << 4) + (v << 1);
            a[2 * v]     = Ps[ln * 2048 + kk + kb];
            a[2 * v + 1] = Ps[ln * 2048 + kk + kb + 1];
        }
#pragma unroll
        for (int j = 0; j < 8; ++j) {
            const int c = wave * 128 + (j << 4) + ln;
            v16bf bfrag;
#pragma unroll
            for (int v = 0; v < 8; ++v) {
                const int kb = (hi << 4) + (v << 1);
                bfrag[2 * v]     = Vs[kb * D_MODEL + c];
                bfrag[2 * v + 1] = Vs[(kb + 1) * D_MODEL + c];
            }
            oacc[j] = WMMA_BF16(a, bfrag, oacc[j]);
        }
#if HAVE_TDM
        if (wave == 0) wait_tensorcnt0();
#endif
        __syncthreads();
    }

    // ---- epilogue: normalize (8 rcp, then muls), write f32 ----
    float rinv[8];
#pragma unroll
    for (int r = 0; r < 8; ++r) rinv[r] = 1.0f / rowsum[(hi << 3) + r];
#pragma unroll
    for (int j = 0; j < 8; ++j) {
        const int c = wave * 128 + (j << 4) + ln;
#pragma unroll
        for (int r = 0; r < 8; ++r) {
            const int m = (hi << 3) + r;
            out[(base + q0 + m) * D_MODEL + c] = oacc[j][r] * rinv[r];
        }
    }
}

// ===========================================================================
// Host launch
// ===========================================================================
extern "C" void kernel_launch(void* const* d_in, const int* in_sizes, int n_in,
                              void* d_out, int out_size, void* d_ws, size_t ws_size,
                              hipStream_t stream)
{
    (void)in_sizes; (void)n_in; (void)out_size; (void)ws_size;

    const float* x  = (const float*)d_in[0];
    const float* Wq = (const float*)d_in[1];
    const float* bq = (const float*)d_in[2];
    const float* Wk = (const float*)d_in[3];
    const float* bk = (const float*)d_in[4];
    const float* Wv = (const float*)d_in[5];
    const float* bv = (const float*)d_in[6];
    float* out = (float*)d_out;

    // workspace (bf16): Q,K,V [8192x1024] + xb [8192x1024] + 3x W [1024x1024]
    __bf16* Qb  = (__bf16*)d_ws;
    __bf16* Kb  = Qb  + (size_t)NROWS * D_MODEL;
    __bf16* Vb  = Kb  + (size_t)NROWS * D_MODEL;
    __bf16* xb  = Vb  + (size_t)NROWS * D_MODEL;
    __bf16* Wqb = xb  + (size_t)NROWS * D_MODEL;
    __bf16* Wkb = Wqb + (size_t)D_MODEL * D_MODEL;
    __bf16* Wvb = Wkb + (size_t)D_MODEL * D_MODEL;

    // Pass 0: one-time f32 -> bf16 conversions
    cvt_bf16_kernel<<<dim3(NROWS * D_MODEL / 2048), dim3(256), 0, stream>>>(x, xb);
    cvt_bf16_kernel<<<dim3(D_MODEL * D_MODEL / 2048), dim3(256), 0, stream>>>(Wq, Wqb);
    cvt_bf16_kernel<<<dim3(D_MODEL * D_MODEL / 2048), dim3(256), 0, stream>>>(Wk, Wkb);
    cvt_bf16_kernel<<<dim3(D_MODEL * D_MODEL / 2048), dim3(256), 0, stream>>>(Wv, Wvb);

    // Pass 1: fused QKV projection
    dim3 g1(NROWS / 128, D_MODEL / 128, 3);
    qkv_proj_kernel<<<g1, dim3(256), 0, stream>>>(xb, Wqb, bq, Wkb, bk, Wvb, bv,
                                                  Qb, Kb, Vb);

    // Pass 2: causal attention (230.5KB dynamic LDS; gfx1250 WGP has 320KB)
    (void)hipFuncSetAttribute(reinterpret_cast<const void*>(attn_kernel),
                              hipFuncAttributeMaxDynamicSharedMemorySize,
                              (int)LDS_TOT);
    dim3 g2(BATCH * (SEQ / 16));
    attn_kernel<<<g2, dim3(256), LDS_TOT, stream>>>(Qb, Kb, Vb, out);
}
